// BA_28784870818370
// MI455X (gfx1250) — matrix operand order repaired and checked by reading, not compiled, not run on hardware
//
#include <hip/hip_runtime.h>
#include <hip/hip_bf16.h>

// ---------------------------------------------------------------------------
// BiFormer-style block (windowed top-k attention + dual VMamba SS2D) for
// gfx1250.  All dense projections run through v_wmma_f32_16x16x32_f16.
// GEMM uses 64x16 register blocking per wave (4 WMMA accumulators sharing one
// B fragment) + global_prefetch of the next K chunk.
// ---------------------------------------------------------------------------

typedef __attribute__((ext_vector_type(16))) _Float16 v16h;
typedef __attribute__((ext_vector_type(8)))  _Float16 v8h;
typedef __attribute__((ext_vector_type(8)))  float    v8f;

#define B_N   4
#define DIMC  64
#define L_IMG 1024     // 32*32
#define ROWS  4096     // B * 32 * 32
#define DI_C  512
#define DM_C  256

// ---------------------------------------------------------------------------
// WMMA GEMM:  C[M,N](f32) = A[M,K](f16, lda) * Bt[N,K](f16, ldb) + bias
// One wave computes a 64x16 strip of C (4 stacked 16x16 tiles along M),
// reusing the B fragment across the 4 WMMAs of each K step.
// M multiple of 64; N multiple of 16; K multiple of 32.
// A fragment: lane&15 = row, elems 0..7 = K(kb..kb+7), 8..15 = K(16+kb..),
//             kb = (lane>>4)*8   (ISA 16-bit A 16x32 layout)
// B fragment: lane&15 = col, same K layout as A (kb = (lane>>4)*8)
// D layout:   VGPR v -> M = tile*16 + v + (lane>=16 ? 8 : 0), N = lane&15
// ---------------------------------------------------------------------------
__global__ __launch_bounds__(32)
void ba_gemm_wmma(const _Float16* __restrict__ A, const _Float16* __restrict__ Bt,
                  const float* __restrict__ bias, float* __restrict__ C,
                  int K, int lda, int ldb, int ldc)
{
    const int lane = threadIdx.x;
    const int tn = blockIdx.x, tm = blockIdx.y;      // tm covers 64 rows of M
    const int r15 = lane & 15;
    const int kbA = (lane >> 4) * 8;
    const int kbB = (lane >> 4) * 8;
    const _Float16* arow0 = A  + (size_t)(tm * 64 + r15) * lda;
    const _Float16* brow  = Bt + (size_t)(tn * 16 + r15) * ldb;
    v8f acc[4];
#pragma unroll
    for (int mt = 0; mt < 4; ++mt)
#pragma unroll
        for (int i = 0; i < 8; ++i) acc[mt][i] = 0.f;

    for (int k0 = 0; k0 < K; k0 += 32) {
        if (k0 + 32 < K) {                            // prefetch next K chunk
            __builtin_prefetch(brow + k0 + 32, 0, 1);
#pragma unroll
            for (int mt = 0; mt < 4; ++mt)
                __builtin_prefetch(arow0 + (size_t)(mt * 16) * lda + k0 + 32, 0, 1);
        }
        v8h b0 = *(const v8h*)(brow + k0 + kbB);
        v8h b1 = *(const v8h*)(brow + k0 + 16 + kbB);
        v16h bv;
#pragma unroll
        for (int i = 0; i < 8; ++i) { bv[i] = b0[i]; bv[8 + i] = b1[i]; }
#pragma unroll
        for (int mt = 0; mt < 4; ++mt) {
            const _Float16* arow = arow0 + (size_t)(mt * 16) * lda;
            v8h a0 = *(const v8h*)(arow + k0 + kbA);
            v8h a1 = *(const v8h*)(arow + k0 + 16 + kbA);
            v16h av;
#pragma unroll
            for (int i = 0; i < 8; ++i) { av[i] = a0[i]; av[8 + i] = a1[i]; }
            acc[mt] = __builtin_amdgcn_wmma_f32_16x16x32_f16(
                /*neg_a=*/false, av, /*neg_b=*/false, bv,
                /*c_mod=*/(short)0, acc[mt], /*reuse_a=*/false, /*reuse_b=*/false);
        }
    }
    const int ccol = tn * 16 + r15;
    const float bb = bias ? bias[ccol] : 0.0f;
#pragma unroll
    for (int mt = 0; mt < 4; ++mt) {
        const int rbase = tm * 64 + mt * 16 + ((lane >> 4) * 8);
#pragma unroll
        for (int v = 0; v < 8; ++v)
            C[(size_t)(rbase + v) * ldc + ccol] = acc[mt][v] + bb;
    }
}

// ---- weight packers: fp32 -> f16 [N][Kpad] ---------------------------------
__global__ void ba_pack_w_kn(const float* __restrict__ src, _Float16* __restrict__ dst,
                             int K, int N, int Kpad)   // src is [K][N]
{
    int idx = blockIdx.x * blockDim.x + threadIdx.x;
    if (idx >= N * Kpad) return;
    int k = idx % Kpad, n = idx / Kpad;
    dst[idx] = (k < K) ? (_Float16)src[(size_t)k * N + n] : (_Float16)0.f;
}
__global__ void ba_pack_w_nk(const float* __restrict__ src, _Float16* __restrict__ dst,
                             int K, int N, int Kpad)   // src is [N][K]
{
    int idx = blockIdx.x * blockDim.x + threadIdx.x;
    if (idx >= N * Kpad) return;
    int k = idx % Kpad, n = idx / Kpad;
    dst[idx] = (k < K) ? (_Float16)src[(size_t)n * K + k] : (_Float16)0.f;
}

// ---- stage A: x (b,64,32,32) -> window-ordered rows f16 [4096][64] ---------
__global__ void ba_pack_xwin(const float* __restrict__ x, _Float16* __restrict__ A)
{
    int idx = blockIdx.x * blockDim.x + threadIdx.x;   // 4096*64
    if (idx >= ROWS * 64) return;
    int c = idx & 63, row = idx >> 6;
    int b = row >> 10, t = row & 1023;
    int n = t >> 4, s = t & 15;
    int h = (n >> 3) * 4 + (s >> 2);
    int w = (n & 7) * 4 + (s & 3);
    A[idx] = (_Float16)x[(((size_t)b * 64 + c) * 32 + h) * 32 + w];
}

// ---- window means of q and k ----------------------------------------------
__global__ void ba_win_means(const float* __restrict__ qkv,
                             float* __restrict__ qwin, float* __restrict__ kwin)
{
    int idx = blockIdx.x * blockDim.x + threadIdx.x;   // 4*64*64
    if (idx >= 4 * 64 * 64) return;
    int c = idx & 63, n = (idx >> 6) & 63, b = idx >> 12;
    float sq = 0.f, sk = 0.f;
    for (int s = 0; s < 16; ++s) {
        const float* r = qkv + (((size_t)b * 64 + n) * 16 + s) * 192;
        sq += r[c]; sk += r[64 + c];
    }
    qwin[idx] = sq * (1.f / 16.f);
    kwin[idx] = sk * (1.f / 16.f);
}

// ---- routing logits + top-4 (descending, ties -> lower index) --------------
__global__ __launch_bounds__(64)
void ba_topk(const float* __restrict__ qwin, const float* __restrict__ kwin,
             int* __restrict__ ridx)
{
    __shared__ float lg[64];
    int blk = blockIdx.x, b = blk >> 6, n = blk & 63;
    int m = threadIdx.x;
    const float* q = qwin + ((size_t)b * 64 + n) * 64;
    const float* k = kwin + ((size_t)b * 64 + m) * 64;
    float dot = 0.f;
    for (int c = 0; c < 64; ++c) dot += q[c] * k[c];
    lg[m] = dot * 0.125f;                       // SCALE = 64^-0.5
    __syncthreads();
    if (m == 0) {
        unsigned long long used = 0ull;
        for (int j = 0; j < 4; ++j) {
            float best = -3.4e38f; int bi = 0;
            for (int mm = 0; mm < 64; ++mm)
                if (!((used >> mm) & 1ull) && lg[mm] > best) { best = lg[mm]; bi = mm; }
            used |= 1ull << bi;
            ridx[((size_t)b * 64 + n) * 4 + j] = bi;
        }
    }
}

// ---- gather selected kv windows into SS2D image f16 [4096][256] ------------
__global__ void ba_build_img(const float* __restrict__ qkv, const int* __restrict__ ridx,
                             _Float16* __restrict__ dst, int choff)
{
    int idx = blockIdx.x * blockDim.x + threadIdx.x;   // 4*1024*256
    if (idx >= B_N * L_IMG * DM_C) return;
    int m = idx & 255, p = (idx >> 8) & 1023, b = idx >> 18;
    int h = p >> 5, w = p & 31;
    int n = (h >> 2) * 8 + (w >> 2);
    int s = (h & 3) * 4 + (w & 3);
    int j = m >> 6, d = m & 63;
    int sw = ridx[((size_t)b * 64 + n) * 4 + j];
    dst[idx] = (_Float16)qkv[(((size_t)b * 64 + sw) * 16 + s) * 192 + choff + d];
}

// ---- depthwise 3x3 conv (SAME) + SiLU;  xz rows (b,p)[1024] -> xc [b][512][1024]
__global__ void ba_dwconv_silu(const float* __restrict__ xz, const float* __restrict__ cw,
                               const float* __restrict__ cb, float* __restrict__ xc)
{
    int idx = blockIdx.x * blockDim.x + threadIdx.x;   // 4*512*1024
    if (idx >= B_N * DI_C * L_IMG) return;
    int p = idx & 1023, d = (idx >> 10) & 511, b = idx >> 19;
    int h = p >> 5, w = p & 31;
    float acc = cb[d];
    for (int kh = 0; kh < 3; ++kh) {
        int hh = h + kh - 1; if (hh < 0 || hh > 31) continue;
        for (int kw = 0; kw < 3; ++kw) {
            int ww = w + kw - 1; if (ww < 0 || ww > 31) continue;
            acc += cw[(kh * 3 + kw) * DI_C + d] *
                   xz[(((size_t)b * L_IMG) + hh * 32 + ww) * 1024 + d];
        }
    }
    xc[((size_t)b * DI_C + d) * L_IMG + p] = acc / (1.f + __expf(-acc));
}

// ---- build 4-direction scan inputs f16 [4][4096][512] ----------------------
__global__ void ba_pack_xs(const float* __restrict__ xc, _Float16* __restrict__ xsA)
{
    int idx = blockIdx.x * blockDim.x + threadIdx.x;   // 4*4096*512 = 2^23
    if (idx >= 4 * ROWS * DI_C) return;
    int d = idx & 511, row = (idx >> 9) & 4095, k = idx >> 21;
    int b = row >> 10, l = row & 1023;
    int ll = (k >= 2) ? (1023 - l) : l;
    int p  = (k & 1) ? ((ll & 31) * 32 + (ll >> 5)) : ll;
    xsA[idx] = (_Float16)xc[((size_t)b * DI_C + d) * L_IMG + p];
}

// ---- dts (first 16 cols of x_dbl) padded K=16 -> 32, f16 -------------------
__global__ void ba_pack_dtA(const float* __restrict__ xdbl, _Float16* __restrict__ dtA)
{
    int idx = blockIdx.x * blockDim.x + threadIdx.x;   // 4*4096*32 = 2^19
    if (idx >= 4 * ROWS * 32) return;
    int r = idx & 31, row = (idx >> 5) & 4095, k = idx >> 17;
    dtA[idx] = (r < 16) ? (_Float16)xdbl[((size_t)k * ROWS + row) * 48 + r]
                        : (_Float16)0.f;
}

// ---- selective scan: one thread per (k,b,d), 1024 sequential steps ---------
__global__ void ba_scan(const _Float16* __restrict__ xsA,   // [4][4096][512]
                        const float* __restrict__ xdbl,     // [4][4096][48]
                        const float* __restrict__ dpre,     // [4][4096][512]
                        const float* __restrict__ A_log,    // [4][512][16]
                        const float* __restrict__ Dp,       // [4][512]
                        float* __restrict__ ybuf)           // [4][4096][512]
{
    int idx = blockIdx.x * blockDim.x + threadIdx.x;
    if (idx >= 4 * 4 * DI_C) return;
    int d = idx & 511, b = (idx >> 9) & 3, k = idx >> 11;
    float Ac[16], h[16];
#pragma unroll
    for (int n = 0; n < 16; ++n) {
        Ac[n] = -__expf(A_log[((size_t)k * DI_C + d) * 16 + n]);
        h[n] = 0.f;
    }
    float Dv = Dp[k * DI_C + d];
    size_t rbase = (size_t)k * ROWS + (size_t)b * L_IMG;
    for (int l = 0; l < 1024; ++l) {
        size_t r = rbase + l;
        float dp  = dpre[r * DI_C + d];
        float dlt = (dp > 20.f) ? dp : log1pf(__expf(dp));   // softplus
        float u   = (float)xsA[r * DI_C + d];
        float du  = dlt * u;
        const float* bc = xdbl + r * 48;
        float y = 0.f;
#pragma unroll
        for (int n = 0; n < 16; ++n) {
            h[n] = h[n] * __expf(dlt * Ac[n]) + du * bc[16 + n];
            y += h[n] * bc[32 + n];
        }
        ybuf[r * DI_C + d] = y + Dv * u;
    }
}

// ---- recombine 4 directions + LayerNorm(512,1e-5) + SiLU(z) gate -> f16 ----
__global__ __launch_bounds__(256)
void ba_ln_gate_pack(const float* __restrict__ ybuf, const float* __restrict__ xz,
                     const float* __restrict__ lnw, const float* __restrict__ lnb,
                     _Float16* __restrict__ outA)
{
    __shared__ float s1[256], s2[256];
    int row = blockIdx.x;                 // b*1024 + p
    int b = row >> 10, p = row & 1023;
    int l1 = (p & 31) * 32 + (p >> 5);
    int t = threadIdx.x;
    float vals[2], lsum = 0.f, lsq = 0.f;
#pragma unroll
    for (int i = 0; i < 2; ++i) {
        int c = t + i * 256;
        float y = ybuf[((size_t)0 * ROWS + b * L_IMG + p)            * DI_C + c]
                + ybuf[((size_t)2 * ROWS + b * L_IMG + (1023 - p))   * DI_C + c]
                + ybuf[((size_t)1 * ROWS + b * L_IMG + l1)           * DI_C + c]
                + ybuf[((size_t)3 * ROWS + b * L_IMG + (1023 - l1))  * DI_C + c];
        vals[i] = y; lsum += y; lsq += y * y;
    }
    s1[t] = lsum; s2[t] = lsq; __syncthreads();
    for (int s = 128; s > 0; s >>= 1) {
        if (t < s) { s1[t] += s1[t + s]; s2[t] += s2[t + s]; }
        __syncthreads();
    }
    float mean = s1[0] * (1.f / 512.f);
    float var  = s2[0] * (1.f / 512.f) - mean * mean;
    float inv  = rsqrtf(var + 1e-5f);
#pragma unroll
    for (int i = 0; i < 2; ++i) {
        int c = t + i * 256;
        float z  = xz[((size_t)b * L_IMG + p) * 1024 + 512 + c];
        float g  = z / (1.f + __expf(-z));
        float yn = (vals[i] - mean) * inv * lnw[c] + lnb[c];
        outA[((size_t)b * L_IMG + p) * DI_C + c] = (_Float16)(yn * g);
    }
}

// ---- attention: per (b,window), 8 heads x 16 queries, 64 keys, dh=8 --------
__global__ __launch_bounds__(128)
void ba_attn(const float* __restrict__ qkv, const float* __restrict__ kimg,
             const float* __restrict__ vimg, _Float16* __restrict__ outA)
{
    __shared__ float kb[64 * 64], vb[64 * 64], qb[16 * 64];
    int blk = blockIdx.x, b = blk >> 6, n = blk & 63;
    int t = threadIdx.x;
    int wy = n >> 3, wx = n & 7;
    for (int i = t; i < 64 * 64; i += 128) {
        int kk = i >> 6, c = i & 63;
        int j = kk >> 4, s = kk & 15;
        int h = wy * 4 + (s >> 2), w = wx * 4 + (s & 3);
        size_t pix = (size_t)b * L_IMG + h * 32 + w;
        kb[i] = kimg[pix * DM_C + j * 64 + c];
        vb[i] = vimg[pix * DM_C + j * 64 + c];
    }
    for (int i = t; i < 16 * 64; i += 128) {
        int qi = i >> 6, c = i & 63;
        qb[i] = qkv[(((size_t)b * 64 + n) * 16 + qi) * 192 + c];
    }
    __syncthreads();
    int hh = t >> 4, qi = t & 15;
    float m = -3.4e38f, ssum = 0.f, o[8];
#pragma unroll
    for (int d = 0; d < 8; ++d) o[d] = 0.f;
    for (int kk = 0; kk < 64; ++kk) {
        float dot = 0.f;
#pragma unroll
        for (int d = 0; d < 8; ++d)
            dot += qb[qi * 64 + hh * 8 + d] * kb[kk * 64 + hh * 8 + d];
        dot *= 0.125f;
        if (dot > m) {                       // online softmax rescale
            float sc = __expf(m - dot);
            ssum *= sc;
#pragma unroll
            for (int d = 0; d < 8; ++d) o[d] *= sc;
            m = dot;
        }
        float e = __expf(dot - m);
        ssum += e;
#pragma unroll
        for (int d = 0; d < 8; ++d) o[d] += e * vb[kk * 64 + hh * 8 + d];
    }
    float inv = 1.f / ssum;
    int h = wy * 4 + (qi >> 2), w = wx * 4 + (qi & 3);
    size_t pix = (size_t)b * L_IMG + h * 32 + w;
#pragma unroll
    for (int d = 0; d < 8; ++d)
        outA[pix * 64 + hh * 8 + d] = (_Float16)(o[d] * inv);
}

// ---- final LayerNorm(64,1e-6) + exact GELU ---------------------------------
__global__ void ba_ln_gelu(const float* __restrict__ in, const float* __restrict__ lnw,
                           const float* __restrict__ lnb, float* __restrict__ out)
{
    int row = blockIdx.x * blockDim.x + threadIdx.x;
    if (row >= ROWS) return;
    const float* x = in + (size_t)row * 64;
    float s = 0.f;
    for (int c = 0; c < 64; ++c) s += x[c];
    float mean = s * (1.f / 64.f);
    float v = 0.f;
    for (int c = 0; c < 64; ++c) { float d = x[c] - mean; v += d * d; }
    float inv = rsqrtf(v * (1.f / 64.f) + 1e-6f);
    for (int c = 0; c < 64; ++c) {
        float y = (x[c] - mean) * inv * lnw[c] + lnb[c];
        out[(size_t)row * 64 + c] = 0.5f * y * (1.f + erff(y * 0.70710678118f));
    }
}

// ---------------------------------------------------------------------------
// Host side
// ---------------------------------------------------------------------------
struct SSP {
    const float *in_proj_w, *conv_w, *conv_b, *x_proj_w, *dt_w, *dt_b;
    const float *A_log, *D, *out_ln_w, *out_ln_b, *out_proj_w;
};

static SSP make_ssp(void* const* d_in, const int* in_sizes, int base)
{
    SSP p;
    if (in_sizes[base] == 262144) {          // dict insertion order
        p.in_proj_w  = (const float*)d_in[base + 0];
        p.conv_w     = (const float*)d_in[base + 1];
        p.conv_b     = (const float*)d_in[base + 2];
        p.x_proj_w   = (const float*)d_in[base + 3];
        p.dt_w       = (const float*)d_in[base + 4];
        p.dt_b       = (const float*)d_in[base + 5];
        p.A_log      = (const float*)d_in[base + 6];
        p.D          = (const float*)d_in[base + 7];
        p.out_ln_w   = (const float*)d_in[base + 8];
        p.out_ln_b   = (const float*)d_in[base + 9];
        p.out_proj_w = (const float*)d_in[base + 10];
    } else {                                 // sorted pytree key order
        p.A_log      = (const float*)d_in[base + 0];
        p.D          = (const float*)d_in[base + 1];
        p.conv_b     = (const float*)d_in[base + 2];
        p.conv_w     = (const float*)d_in[base + 3];
        p.dt_b       = (const float*)d_in[base + 4];
        p.dt_w       = (const float*)d_in[base + 5];
        p.in_proj_w  = (const float*)d_in[base + 6];
        p.out_ln_b   = (const float*)d_in[base + 7];
        p.out_ln_w   = (const float*)d_in[base + 8];
        p.out_proj_w = (const float*)d_in[base + 9];
        p.x_proj_w   = (const float*)d_in[base + 10];
    }
    return p;
}

static inline void* wsp(void* base, size_t& off, size_t bytes)
{
    off = (off + 255) & ~(size_t)255;
    void* p = (char*)base + off;
    off += bytes;
    return p;
}

static void run_ss2d(const SSP& p, const float* qkv, const int* ridx, int choff,
                     float* branch_out,
                     _Float16* img, _Float16* inprojT, float* xz, float* xc,
                     _Float16* xprojT, _Float16* xsA, float* xdbl,
                     _Float16* dtwT, _Float16* dtA, float* dpre, float* ybuf,
                     _Float16* gated, _Float16* outprojT, hipStream_t stream)
{
    const int T = 256;
    ba_build_img<<<(B_N * L_IMG * DM_C + T - 1) / T, T, 0, stream>>>(qkv, ridx, img, choff);

    // in_proj: [4096,256] x [256,1024]
    ba_pack_w_kn<<<(1024 * 256 + T - 1) / T, T, 0, stream>>>(p.in_proj_w, inprojT, 256, 1024, 256);
    { dim3 g(1024 / 16, ROWS / 64);
      ba_gemm_wmma<<<g, 32, 0, stream>>>(img, inprojT, (const float*)nullptr, xz, 256, 256, 256, 1024); }

    ba_dwconv_silu<<<(B_N * DI_C * L_IMG + T - 1) / T, T, 0, stream>>>(xz, p.conv_w, p.conv_b, xc);
    ba_pack_xs<<<(4 * ROWS * DI_C + T - 1) / T, T, 0, stream>>>(xc, xsA);

    // x_proj per direction: [4096,512] x [512,48]   (weight stored [k][48][512])
    ba_pack_w_nk<<<(4 * 48 * 512 + T - 1) / T, T, 0, stream>>>(p.x_proj_w, xprojT, 512, 4 * 48, 512);
    for (int k = 0; k < 4; ++k) {
        dim3 g(48 / 16, ROWS / 64);
        ba_gemm_wmma<<<g, 32, 0, stream>>>(xsA + (size_t)k * ROWS * DI_C,
                                           xprojT + (size_t)k * 48 * 512,
                                           (const float*)nullptr,
                                           xdbl + (size_t)k * ROWS * 48,
                                           512, 512, 512, 48);
    }

    // dt projection per direction: [4096,32(pad16)] x [32,512] + dt_b
    ba_pack_w_nk<<<(4 * 512 * 32 + T - 1) / T, T, 0, stream>>>(p.dt_w, dtwT, 16, 4 * 512, 32);
    ba_pack_dtA<<<(4 * ROWS * 32 + T - 1) / T, T, 0, stream>>>(xdbl, dtA);
    for (int k = 0; k < 4; ++k) {
        dim3 g(512 / 16, ROWS / 64);
        ba_gemm_wmma<<<g, 32, 0, stream>>>(dtA + (size_t)k * ROWS * 32,
                                           dtwT + (size_t)k * 512 * 32,
                                           p.dt_b + k * 512,
                                           dpre + (size_t)k * ROWS * DI_C,
                                           32, 32, 32, 512);
    }

    ba_scan<<<(4 * 4 * DI_C + 255) / 256, 256, 0, stream>>>(xsA, xdbl, dpre, p.A_log, p.D, ybuf);
    ba_ln_gate_pack<<<ROWS, 256, 0, stream>>>(ybuf, xz, p.out_ln_w, p.out_ln_b, gated);

    // out_proj: [4096,512] x [512,256]
    ba_pack_w_kn<<<(256 * 512 + T - 1) / T, T, 0, stream>>>(p.out_proj_w, outprojT, 512, 256, 512);
    { dim3 g(256 / 16, ROWS / 64);
      ba_gemm_wmma<<<g, 32, 0, stream>>>(gated, outprojT, (const float*)nullptr,
                                         branch_out, 512, 512, 512, 256); }
}

extern "C" void kernel_launch(void* const* d_in, const int* in_sizes, int n_in,
                              void* d_out, int out_size, void* d_ws, size_t ws_size,
                              hipStream_t stream)
{
    (void)n_in; (void)out_size; (void)ws_size;
    const float* x      = (const float*)d_in[0];
    const float* w_qkv  = (const float*)d_in[1];
    const float* b_qkv  = (const float*)d_in[2];
    const float* w_mlp  = (const float*)d_in[3];
    const float* b_mlp  = (const float*)d_in[4];
    const float* ln_w   = (const float*)d_in[5];
    const float* ln_b   = (const float*)d_in[6];
    SSP pv = make_ssp(d_in, in_sizes, 7);
    SSP pk = make_ssp(d_in, in_sizes, 18);

    size_t off = 0;
    _Float16* Awin     = (_Float16*)wsp(d_ws, off, (size_t)ROWS * 64 * 2);
    _Float16* wqkvT    = (_Float16*)wsp(d_ws, off, 192 * 64 * 2);
    float*    qkv      = (float*)   wsp(d_ws, off, (size_t)ROWS * 192 * 4);
    float*    qwin     = (float*)   wsp(d_ws, off, 4 * 64 * 64 * 4);
    float*    kwin     = (float*)   wsp(d_ws, off, 4 * 64 * 64 * 4);
    int*      ridx     = (int*)     wsp(d_ws, off, 4 * 64 * 4 * 4);
    _Float16* img      = (_Float16*)wsp(d_ws, off, (size_t)ROWS * 256 * 2);
    _Float16* inprojT  = (_Float16*)wsp(d_ws, off, 1024 * 256 * 2);
    float*    xz       = (float*)   wsp(d_ws, off, (size_t)ROWS * 1024 * 4);
    float*    xc       = (float*)   wsp(d_ws, off, (size_t)B_N * DI_C * L_IMG * 4);
    _Float16* xprojT   = (_Float16*)wsp(d_ws, off, (size_t)4 * 48 * 512 * 2);
    _Float16* xsA      = (_Float16*)wsp(d_ws, off, (size_t)4 * ROWS * DI_C * 2);
    float*    xdbl     = (float*)   wsp(d_ws, off, (size_t)4 * ROWS * 48 * 4);
    _Float16* dtwT     = (_Float16*)wsp(d_ws, off, (size_t)4 * 512 * 32 * 2);
    _Float16* dtA      = (_Float16*)wsp(d_ws, off, (size_t)4 * ROWS * 32 * 2);
    float*    dpre     = (float*)   wsp(d_ws, off, (size_t)4 * ROWS * DI_C * 4);
    float*    ybuf     = (float*)   wsp(d_ws, off, (size_t)4 * ROWS * DI_C * 4);
    _Float16* gated    = (_Float16*)wsp(d_ws, off, (size_t)ROWS * DI_C * 2);
    _Float16* outprojT = (_Float16*)wsp(d_ws, off, 256 * 512 * 2);
    float*    vout     = (float*)   wsp(d_ws, off, (size_t)ROWS * 256 * 4);
    float*    kout     = (float*)   wsp(d_ws, off, (size_t)ROWS * 256 * 4);
    _Float16* attnA    = (_Float16*)wsp(d_ws, off, (size_t)ROWS * 64 * 2);
    _Float16* wmlpT    = (_Float16*)wsp(d_ws, off, 64 * 64 * 2);
    float*    mlpo     = (float*)   wsp(d_ws, off, (size_t)ROWS * 64 * 4);

    const int T = 256;
    // Stage A: qkv projection + routing
    ba_pack_xwin<<<(ROWS * 64 + T - 1) / T, T, 0, stream>>>(x, Awin);
    ba_pack_w_kn<<<(192 * 64 + T - 1) / T, T, 0, stream>>>(w_qkv, wqkvT, 64, 192, 64);
    { dim3 g(192 / 16, ROWS / 64);
      ba_gemm_wmma<<<g, 32, 0, stream>>>(Awin, wqkvT, b_qkv, qkv, 64, 64, 64, 192); }
    ba_win_means<<<(4 * 64 * 64 + T - 1) / T, T, 0, stream>>>(qkv, qwin, kwin);
    ba_topk<<<4 * 64, 64, 0, stream>>>(qwin, kwin, ridx);

    // SS2D on v (channels 128..191) and k (channels 64..127); scratch reused
    run_ss2d(pv, qkv, ridx, 128, vout, img, inprojT, xz, xc, xprojT, xsA, xdbl,
             dtwT, dtA, dpre, ybuf, gated, outprojT, stream);
    run_ss2d(pk, qkv, ridx, 64, kout, img, inprojT, xz, xc, xprojT, xsA, xdbl,
             dtwT, dtA, dpre, ybuf, gated, outprojT, stream);

    // attention + MLP + final LN/GELU
    ba_attn<<<4 * 64, 128, 0, stream>>>(qkv, kout, vout, attnA);
    ba_pack_w_kn<<<(64 * 64 + T - 1) / T, T, 0, stream>>>(w_mlp, wmlpT, 64, 64, 64);
    { dim3 g(64 / 16, ROWS / 64);
      ba_gemm_wmma<<<g, 32, 0, stream>>>(attnA, wmlpT, b_mlp, mlpo, 64, 64, 64, 64); }
    ba_ln_gelu<<<(ROWS + T - 1) / T, T, 0, stream>>>(mlpo, ln_w, ln_b, (float*)d_out);
}
